// CGATLayer_74302934220879
// MI455X (gfx1250) — matrix operand order repaired and checked by reading, not compiled, not run on hardware
//
#include <hip/hip_runtime.h>
#include <hip/hip_bf16.h>

#define NN 20000
#define EE 320000
#define BB 2
#define FF 64
#define RR 32
#define BF 128          // B*F
#define ROWS 40000      // N*B
#define KTOP 160000     // E * 0.5
#define NEG_SLOPE 0.2f
#define LN_EPS 1e-5f

typedef __attribute__((ext_vector_type(16))) _Float16 v16h;
typedef __attribute__((ext_vector_type(8)))  float    v8f;

// ---------------------------------------------------------------- utilities
__device__ inline void atomicMaxF(float* addr, float v) {
    if (v >= 0.0f) atomicMax((int*)addr, __float_as_int(v));
    else           atomicMin((unsigned int*)addr, __float_as_uint(v));
}

// ------------------------------------------------ one-time: layer_w -> f16
__global__ __launch_bounds__(256) void k_convW(const float* __restrict__ w,
                                               _Float16* __restrict__ wh) {
    int i = blockIdx.x * 256 + threadIdx.x;
    if (i < FF * BF) wh[i] = (_Float16)w[i];
}

// ------------------------------------------------ per-iter init
__global__ __launch_bounds__(256) void k_init(float* __restrict__ nmax,
                                              float* __restrict__ denom,
                                              float* __restrict__ colsum) {
    int i = blockIdx.x * 256 + threadIdx.x;
    if (i < NN) { nmax[i] = -3.0e38f; denom[i] = 0.0f; }
    if (blockIdx.x == 0 && threadIdx.x < BF) colsum[threadIdx.x] = 0.0f;
}

// ------------------------------------------------ column sum of x (for readout)
__global__ __launch_bounds__(128) void k_colsum(const float* __restrict__ x,
                                                float* __restrict__ colsum) {
    const int t = threadIdx.x;           // column 0..127
    const int n0 = blockIdx.x * 200;
    float s = 0.0f;
    for (int n = n0; n < n0 + 200; ++n) s += x[n * BF + t];
    atomicAdd(&colsum[t], s);
}

// ------------------------------------------------ tiny per-iter precompute
// wa = W @ a[iter]            [128]
// rel[r][c] (flattened reshape of query@rel_w.T + rel_b)   [32][128]
// rdot[r] = rel[r]·wa         [32]
// trans = (colsum/N) @ tr_w.T + tr_b                        [128]
__global__ __launch_bounds__(128) void k_setup(
    const float* __restrict__ query, const float* __restrict__ rel_w,
    const float* __restrict__ rel_b, const float* __restrict__ Wm,
    const float* __restrict__ avec, int iter,
    const float* __restrict__ tr_w, const float* __restrict__ tr_b,
    const float* __restrict__ colsum,
    float* __restrict__ wa, float* __restrict__ rel,
    float* __restrict__ rdot, float* __restrict__ trans) {
    const int t = threadIdx.x;
    // wa
    {
        float s = 0.0f;
        for (int j = 0; j < FF; ++j) s += Wm[t * FF + j] * avec[iter * FF + j];
        wa[t] = s;
    }
    // rel: y[b][o] = sum_f query[b][f]*rel_w[o][f] + rel_b[o]
    // rel[r][c] = y[r>>4][(r&15)*128 + c]
    for (int idx = t; idx < RR * BF; idx += 128) {
        int r = idx >> 7, c = idx & 127;
        int b = r >> 4;
        int o = (r & 15) * 128 + c;
        float s = rel_b[o];
        for (int f = 0; f < FF; ++f) s += query[b * FF + f] * rel_w[o * FF + f];
        rel[idx] = s;
    }
    __syncthreads();
    if (t < RR) {
        float s = 0.0f;
        for (int c = 0; c < BF; ++c) s += rel[t * BF + c] * wa[c];
        rdot[t] = s;
    }
    {
        int b = t >> 6, fo = t & 63;
        float s = tr_b[fo];
        const float invN = 1.0f / (float)NN;
        for (int f = 0; f < FF; ++f)
            s += (colsum[b * FF + f] * invN) * tr_w[fo * FF + f];
        trans[t] = s;
    }
}

// ------------------------------------------------ per-node dot: xdot[n]=x[n]·wa
__global__ __launch_bounds__(256) void k_xdot(const float* __restrict__ x,
                                              const float* __restrict__ wa,
                                              float* __restrict__ xdot) {
    const int wave = threadIdx.x >> 5, lane = threadIdx.x & 31;
    const int n = blockIdx.x * 8 + wave;
    if (n >= NN) return;
    float4 xv = ((const float4*)(x + n * BF))[lane];
    float4 wv = ((const float4*)wa)[lane];
    float d = xv.x * wv.x + xv.y * wv.y + xv.z * wv.z + xv.w * wv.w;
    for (int m = 16; m >= 1; m >>= 1) d += __shfl_xor(d, m, 32);
    if (lane == 0) xdot[n] = d;
}

// ------------------------------------------------ edge pass 1: score + seg-max
__global__ __launch_bounds__(256) void k_edge1(const int* __restrict__ src,
                                               const int* __restrict__ ridx,
                                               const float* __restrict__ xdot,
                                               const float* __restrict__ rdot,
                                               float* __restrict__ aPre,
                                               float* __restrict__ nmax) {
    int e = blockIdx.x * 256 + threadIdx.x;
    if (e >= EE) return;
    int s = src[e];
    float ap = xdot[s] + rdot[ridx[e]];
    ap = ap > 0.0f ? ap : NEG_SLOPE * ap;
    aPre[e] = ap;
    atomicMaxF(&nmax[s], ap);
}

// ------------------------------------------------ edge pass 2: exp + seg-sum
__global__ __launch_bounds__(256) void k_edge2(const int* __restrict__ src,
                                               const float* __restrict__ aPre,
                                               const float* __restrict__ nmax,
                                               float* __restrict__ ex,
                                               float* __restrict__ denom) {
    int e = blockIdx.x * 256 + threadIdx.x;
    if (e >= EE) return;
    int s = src[e];
    float v = __expf(aPre[e] - nmax[s]);
    ex[e] = v;
    atomicAdd(&denom[s], v);
}

// ------------------------------------------------ edge pass 3: normalize + hist
__global__ __launch_bounds__(256) void k_edge3(const int* __restrict__ src,
                                               const float* __restrict__ ex,
                                               const float* __restrict__ denom,
                                               float* __restrict__ alpha,
                                               unsigned int* __restrict__ hist) {
    int e = blockIdx.x * 256 + threadIdx.x;
    if (e >= EE) return;
    float a = ex[e] / (denom[src[e]] + 1e-16f);
    alpha[e] = a;
    atomicAdd(&hist[__float_as_uint(a) >> 16], 1u);
}

// ------------------------------------------------ radix-select threshold (1 block)
__global__ __launch_bounds__(256) void k_thresh(const unsigned int* __restrict__ hist,
                                                unsigned int* __restrict__ thresh) {
    __shared__ unsigned int s[256];
    __shared__ unsigned int carry;
    __shared__ int done;
    if (threadIdx.x == 0) { carry = 0u; done = 0; }
    __syncthreads();
    for (int chunk = 255; chunk >= 0; --chunk) {
        if (done) break;
        int bin = chunk * 256 + (255 - (int)threadIdx.x);  // tid asc == bin desc
        unsigned int cnt = hist[bin];
        s[threadIdx.x] = cnt;
        __syncthreads();
        for (int off = 1; off < 256; off <<= 1) {
            unsigned int v = (threadIdx.x >= (unsigned)off) ? s[threadIdx.x - off] : 0u;
            __syncthreads();
            s[threadIdx.x] += v;
            __syncthreads();
        }
        unsigned int incl = s[threadIdx.x];
        unsigned int excl = incl - cnt;
        unsigned int base = carry;
        if (base + excl < (unsigned)KTOP && base + incl >= (unsigned)KTOP) {
            *thresh = ((unsigned int)bin) << 16;
            done = 1;
        }
        __syncthreads();
        if (threadIdx.x == 255 && !done) carry = base + incl;
        __syncthreads();
    }
    if (threadIdx.x == 0 && !done) *thresh = 0u;
}

// ------------------------------------------------ masked weighted scatter (wave/edge)
__global__ __launch_bounds__(256) void k_scatter(const int* __restrict__ src,
                                                 const int* __restrict__ dst,
                                                 const int* __restrict__ ridx,
                                                 const float* __restrict__ alpha,
                                                 const unsigned int* __restrict__ thresh,
                                                 const float* __restrict__ x,
                                                 const float* __restrict__ rel,
                                                 float* __restrict__ upd) {
    const int wave = threadIdx.x >> 5, lane = threadIdx.x & 31;
    const int e = blockIdx.x * 8 + wave;
    if (e >= EE) return;
    float a = alpha[e];
    if (__float_as_uint(a) < *thresh) return;   // wave-uniform
    int s = src[e], d = dst[e], r = ridx[e];
    float4 xv = ((const float4*)(x + s * BF))[lane];
    float4 rv = ((const float4*)(rel + r * BF))[lane];
    float* up = upd + d * BF + lane * 4;
    atomicAdd(up + 0, a * (xv.x + rv.x));
    atomicAdd(up + 1, a * (xv.y + rv.y));
    atomicAdd(up + 2, a * (xv.z + rv.z));
    atomicAdd(up + 3, a * (xv.w + rv.w));
}

// ------------------------------------------------ WMMA GEMM + LN + ELU + residual
// out[row, :64] = concat(x[row], upd[row]) @ layer_w^T ; row = n*2+b, K=128, N=64
__global__ __launch_bounds__(256) void k_gemm(
    const float* __restrict__ xin, const float* __restrict__ upd,
    const _Float16* __restrict__ wh, const float* __restrict__ lbias,
    const float* __restrict__ trans, const float* __restrict__ lgam,
    const float* __restrict__ lbet, float* __restrict__ xout) {
    __shared__ _Float16 sW[FF * BF];   // 16 KB, layer_w[j][k] in f16
    for (int i = threadIdx.x; i < FF * BF; i += 256) sW[i] = wh[i];
    __syncthreads();

    const int wave = threadIdx.x >> 5, lane = threadIdx.x & 31;
    const int tile = blockIdx.x * 8 + wave;
    if (tile >= ROWS / 16) return;

    const int cl = lane & 15;
    const int hi = lane >> 4;               // 0 / 1 half of the wave
    const int rowA = tile * 16 + cl;        // A-fragment row for this lane

    v8f acc[4] = {};                        // 4 N-tiles of 16 cols

#pragma unroll
    for (int kk = 0; kk < 4; ++kk) {        // K = 4 x 32
        const float* srcp = (kk < 2) ? xin : upd;   // cols 0-63 : x, 64-127 : upd
        const int cb = (kk & 1) * 32;
        const float* p = srcp + rowA * FF;
        const int c0 = cb + hi * 16;
        const int c1 = cb + hi * 16 + 8;
        float4 f0 = *(const float4*)(p + c0);
        float4 f1 = *(const float4*)(p + c0 + 4);
        float4 f2 = *(const float4*)(p + c1);
        float4 f3 = *(const float4*)(p + c1 + 4);
        v16h A;
        A[0] = (_Float16)f0.x; A[1] = (_Float16)f0.y;
        A[2] = (_Float16)f0.z; A[3] = (_Float16)f0.w;
        A[4] = (_Float16)f1.x; A[5] = (_Float16)f1.y;
        A[6] = (_Float16)f1.z; A[7] = (_Float16)f1.w;
        A[8]  = (_Float16)f2.x; A[9]  = (_Float16)f2.y;
        A[10] = (_Float16)f2.z; A[11] = (_Float16)f2.w;
        A[12] = (_Float16)f3.x; A[13] = (_Float16)f3.y;
        A[14] = (_Float16)f3.z; A[15] = (_Float16)f3.w;

        const int koff = kk * 32 + hi * 16;
#pragma unroll
        for (int t = 0; t < 4; ++t) {
            int j = t * 16 + cl;            // output column owned by this lane
            v16h Bm = *(const v16h*)(sW + j * BF + koff);
            acc[t] = __builtin_amdgcn_wmma_f32_16x16x32_f16(
                false, A, false, Bm, (short)0, acc[t], false, false);
        }
    }

    // epilogue: + layer_b + trans[b], LayerNorm over 64, ELU, + x residual
#pragma unroll
    for (int r = 0; r < 8; ++r) {
        const int m = r + hi * 8;           // row-in-tile for this lane half
        const int row = tile * 16 + m;
        const int b = row & 1;
        float v[4];
        float s1 = 0.0f, s2 = 0.0f;
#pragma unroll
        for (int t = 0; t < 4; ++t) {
            int col = t * 16 + cl;
            float val = acc[t][r] + lbias[col] + trans[b * FF + col];
            v[t] = val;
            s1 += val;
            s2 += val * val;
        }
        // reduce across the 16 lanes of this half (xor masks stay in-half)
        for (int msk = 1; msk < 16; msk <<= 1) {
            s1 += __shfl_xor(s1, msk, 32);
            s2 += __shfl_xor(s2, msk, 32);
        }
        float mean = s1 * (1.0f / 64.0f);
        float var = s2 * (1.0f / 64.0f) - mean * mean;
        float rstd = rsqrtf(var + LN_EPS);
#pragma unroll
        for (int t = 0; t < 4; ++t) {
            int col = t * 16 + cl;
            float o = (v[t] - mean) * rstd * lgam[col] + lbet[col];
            o = o > 0.0f ? o : (__expf(o) - 1.0f);          // ELU
            xout[row * FF + col] = o + xin[row * FF + col]; // residual x = h + x
        }
    }
}

// ================================================================ launch
extern "C" void kernel_launch(void* const* d_in, const int* in_sizes, int n_in,
                              void* d_out, int out_size, void* d_ws, size_t ws_size,
                              hipStream_t stream) {
    const int*   eidx    = (const int*)d_in[0];
    const int*   src     = eidx;
    const int*   dst     = eidx + EE;
    const int*   ridx    = (const int*)d_in[1];
    const float* x0      = (const float*)d_in[2];
    const float* query   = (const float*)d_in[3];
    const float* rel_w   = (const float*)d_in[4];
    const float* rel_b   = (const float*)d_in[5];
    const float* layer_w = (const float*)d_in[6];
    const float* layer_b = (const float*)d_in[7];
    const float* tr_w    = (const float*)d_in[8];
    const float* tr_b    = (const float*)d_in[9];
    const float* Wm      = (const float*)d_in[10];
    const float* avec    = (const float*)d_in[11];
    const float* ln_g    = (const float*)d_in[12];
    const float* ln_b    = (const float*)d_in[13];

    char* wsp = (char*)d_ws;
    size_t off = 0;
    auto alloc = [&](size_t bytes) -> void* {
        void* p = wsp + off;
        off += (bytes + 255) & ~(size_t)255;
        return p;
    };
    const size_t xBytes = (size_t)NN * BF * sizeof(float);      // 10.24 MB
    float*        xA     = (float*)alloc(xBytes);
    float*        xB     = (float*)alloc(xBytes);
    float*        upd    = (float*)alloc(xBytes);
    float*        aPre   = (float*)alloc((size_t)EE * 4);       // scores, then alpha
    float*        aVal   = (float*)alloc((size_t)EE * 4);       // exp values
    float*        xdot   = (float*)alloc((size_t)NN * 4);
    float*        nmax   = (float*)alloc((size_t)NN * 4);
    float*        denom  = (float*)alloc((size_t)NN * 4);
    unsigned int* hist   = (unsigned int*)alloc(65536 * 4);
    float*        rel    = (float*)alloc(RR * BF * 4);
    float*        wa     = (float*)alloc(BF * 4);
    float*        rdot   = (float*)alloc(RR * 4);
    float*        colsum = (float*)alloc(BF * 4);
    float*        trans  = (float*)alloc(BF * 4);
    _Float16*     wh     = (_Float16*)alloc(FF * BF * 2);
    unsigned int* thr    = (unsigned int*)alloc(256);

    // one-time: x <- boundary, layer_w -> f16
    hipMemcpyAsync(xA, x0, xBytes, hipMemcpyDeviceToDevice, stream);
    k_convW<<<(FF * BF + 255) / 256, 256, 0, stream>>>(layer_w, wh);

    for (int it = 0; it < 4; ++it) {
        float* xin  = (it & 1) ? xB : xA;
        float* xout = (it == 3) ? (float*)d_out : ((it & 1) ? xA : xB);

        hipMemsetAsync(hist, 0, 65536 * 4, stream);
        k_init<<<(NN + 255) / 256, 256, 0, stream>>>(nmax, denom, colsum);
        hipMemcpyAsync(upd, xin, xBytes, hipMemcpyDeviceToDevice, stream);

        k_colsum<<<NN / 200, 128, 0, stream>>>(xin, colsum);
        k_setup<<<1, 128, 0, stream>>>(query, rel_w, rel_b, Wm, avec, it,
                                       tr_w, tr_b, colsum, wa, rel, rdot, trans);
        k_xdot<<<NN / 8, 256, 0, stream>>>(xin, wa, xdot);
        k_edge1<<<(EE + 255) / 256, 256, 0, stream>>>(src, ridx, xdot, rdot, aPre, nmax);
        k_edge2<<<(EE + 255) / 256, 256, 0, stream>>>(src, aPre, nmax, aVal, denom);
        k_edge3<<<(EE + 255) / 256, 256, 0, stream>>>(src, aVal, denom, aPre, hist);
        k_thresh<<<1, 256, 0, stream>>>(hist, thr);
        k_scatter<<<EE / 8, 256, 0, stream>>>(src, dst, ridx, aPre, thr, xin, rel, upd);
        k_gemm<<<(ROWS / 16 + 7) / 8, 256, 0, stream>>>(xin, upd, wh, layer_b,
                                                        trans, ln_g, ln_b, xout);
    }
}
